// DummyLinear_79379585565207
// MI455X (gfx1250) — compile-verified, matrix operand order checked by reading, not run on hardware
//
#include <hip/hip_runtime.h>
#include <math.h>

typedef __attribute__((ext_vector_type(8))) int   v8i;
typedef __attribute__((ext_vector_type(2))) float f2;

#define ACT_INV    64.0f             /* 1 / 2^-6  */
#define W_INV      128.0f            /* 1 / 2^-7  */
#define BIAS_SCALE 1.220703125e-4f   /* 2^-13 = act_scale * weight_scale */
#define BIAS_INV   8192.0f           /* 1 / 2^-13 */

__device__ __forceinline__ int quant8(float v, float inv) {
    // round-to-nearest-even (matches jnp.round under RNE), clamp to int8, low byte
    float r = __builtin_rintf(v * inv);
    r = __builtin_fmaxf(-128.0f, __builtin_fminf(r, 127.0f));
    return ((int)r) & 0xff;
}

// A-matrix fragment for 16 rows of x (K=10, zero-padded to 64).
// ISA 8-bit A 16x64 layout: VGPR0 lo-lanes=K0..3 / hi-lanes=K8..11,
// VGPR1 lo-lanes=K4..7 / hi-lanes=K12..15; VGPR2..7 = 0.
__device__ __forceinline__ v8i load_a(const float* __restrict__ xr, bool hi) {
    v8i a = {};
    if (!hi) {
        f2 p0 = *(const f2*)(xr + 0);
        f2 p1 = *(const f2*)(xr + 2);
        f2 p2 = *(const f2*)(xr + 4);
        f2 p3 = *(const f2*)(xr + 6);
        a.x = quant8(p0.x, ACT_INV)        | (quant8(p0.y, ACT_INV) << 8) |
              (quant8(p1.x, ACT_INV) << 16) | (quant8(p1.y, ACT_INV) << 24);
        a.y = quant8(p2.x, ACT_INV)        | (quant8(p2.y, ACT_INV) << 8) |
              (quant8(p3.x, ACT_INV) << 16) | (quant8(p3.y, ACT_INV) << 24);
    } else {
        f2 p4 = *(const f2*)(xr + 8);
        a.x = quant8(p4.x, ACT_INV)        | (quant8(p4.y, ACT_INV) << 8);
    }
    return a;
}

__global__ __launch_bounds__(256) void qlinear_iu8_wmma(
    const float* __restrict__ x,   // [nrows, 10]
    const float* __restrict__ W,   // [3, 10]
    const float* __restrict__ b,   // [3]
    float* __restrict__ y,         // [nrows, 3]
    int nrows, int tilesPerWave)
{
    const int  lane = threadIdx.x & 31;
    const int  col  = lane & 15;                 // N index / M-row selector
    const bool hi   = lane >= 16;
    const bool isOutLane = (col < 3);
    const long wave = (long)blockIdx.x * (blockDim.x >> 5) + (threadIdx.x >> 5);

    // ---- B matrix (weights), built once per wave ----------------------------
    // 8-bit B 64x16 layout mirrors A with N across lanes; only N=0..2 nonzero.
    v8i bm = {};
    if (isOutLane) {
        const float* wr = W + col * 10;
        if (!hi) {
            bm.x = quant8(wr[0], W_INV)        | (quant8(wr[1], W_INV) << 8) |
                   (quant8(wr[2], W_INV) << 16) | (quant8(wr[3], W_INV) << 24);
            bm.y = quant8(wr[4], W_INV)        | (quant8(wr[5], W_INV) << 8) |
                   (quant8(wr[6], W_INV) << 16) | (quant8(wr[7], W_INV) << 24);
        } else {
            bm.x = quant8(wr[8], W_INV)        | (quant8(wr[9], W_INV) << 8);
        }
    }

    // ---- bias: fake_quant(b, 2^-13) (saturates to +/-128*2^-13) -------------
    float bias = 0.0f;
    if (isOutLane) {
        float r = __builtin_rintf(b[col] * BIAS_INV);
        r = __builtin_fmaxf(-128.0f, __builtin_fminf(r, 127.0f));
        bias = r * BIAS_SCALE;
    }

    const long tile0    = wave * tilesPerWave;
    const long rowBase0 = tile0 * 16;
    // per-lane base pointers, strength-reduced across the tile loop
    const float* xr = x + (rowBase0 + col) * 10;           // 8B-aligned (40*row)
    float*       yp = y + (rowBase0 + (hi ? 8 : 0)) * 3 + col;

    for (int t = 0; t < tilesPerWave; ++t) {
        const long rowBase = rowBase0 + (long)t * 16;      // uniform per wave
        if (rowBase >= nrows) break;                       // uniform branch

        const v8i c0 = {};
        if (rowBase + 16 <= (long)nrows) {
            // ================= fast path: full tile, no per-row guards =======
            v8i a = load_a(xr, hi);
            // D(i32)[16x16] = A[16x64] x B[64x16], signed int8 (exact)
            v8i d = __builtin_amdgcn_wmma_i32_16x16x64_iu8(true, a, true, bm,
                                                           c0, false, false);
            // D layout: VGPR v -> (M=v, N=lane) lo-lanes, (M=v+8) hi-lanes
            if (isOutLane) {
#pragma unroll
                for (int i = 0; i < 8; ++i)
                    yp[3 * i] = (float)d[i] * BIAS_SCALE + bias;
            }
        } else {
            // ================= tail path: clamp loads, guard stores ==========
            long row = rowBase + col;
            if (row >= nrows) row = nrows - 1;
            v8i a = load_a(x + row * 10, hi);
            v8i d = __builtin_amdgcn_wmma_i32_16x16x64_iu8(true, a, true, bm,
                                                           c0, false, false);
            if (isOutLane) {
                const long r0 = rowBase + (hi ? 8 : 0);
#pragma unroll
                for (int i = 0; i < 8; ++i) {
                    const long rr = r0 + i;
                    if (rr < nrows)
                        y[rr * 3 + col] = (float)d[i] * BIAS_SCALE + bias;
                }
            }
        }
        xr += 160;   // 16 rows * 10 floats
        yp += 48;    // 16 rows * 3 floats
    }
}

extern "C" void kernel_launch(void* const* d_in, const int* in_sizes, int n_in,
                              void* d_out, int out_size, void* d_ws, size_t ws_size,
                              hipStream_t stream) {
    const float* x = (const float*)d_in[0];
    const float* W = (const float*)d_in[1];
    const float* b = (const float*)d_in[2];
    float* y = (float*)d_out;

    const int nrows = in_sizes[0] / 10;             // 4,194,304
    const int tilesPerWave = 8;                     // 128 rows per wave
    const int tiles = (nrows + 15) / 16;
    const int waves = (tiles + tilesPerWave - 1) / tilesPerWave;
    const int wavesPerBlock = 8;                    // 256 threads = 8 wave32
    const int blocks = (waves + wavesPerBlock - 1) / wavesPerBlock;

    qlinear_iu8_wmma<<<blocks, 256, 0, stream>>>(x, W, b, y, nrows, tilesPerWave);
}
